// SCAF_28836410425458
// MI455X (gfx1250) — matrix operand order repaired and checked by reading, not compiled, not run on hardware
//
#include <hip/hip_runtime.h>

// ---------------- problem constants ----------------
#define Bsz 4
#define Cc  64
#define Nn  4096
#define Gg  32
#define Rr  4
#define SZQ (Bsz*Nn*Cc)   // elements per q/k/v buffer (1048576)

typedef __attribute__((ext_vector_type(16))) __bf16   v16bf;
typedef __attribute__((ext_vector_type(8)))  float    v8f;
typedef __attribute__((ext_vector_type(4)))  unsigned int uint4v;
typedef __attribute__((ext_vector_type(4)))  float    f4v;

union Frag16 { v16bf v; uint4v q[2]; unsigned int u[8]; };

// native bf16 convert (hardware cvt on gfx1250)
__device__ __forceinline__ unsigned short f2bf(float x) {
    __bf16 h = (__bf16)x;
    return __builtin_bit_cast(unsigned short, h);
}
__device__ __forceinline__ unsigned int f2bf2(float lo, float hi) {
    return (unsigned int)f2bf(lo) | ((unsigned int)f2bf(hi) << 16);
}
__device__ __forceinline__ float sigmoidf_(float x) {
    return 1.0f / (1.0f + __expf(-x));
}

// async global->LDS copy of 16 bytes (gfx1250 GLOBAL_LOAD_ASYNC_TO_LDS_B128, GV mode)
__device__ __forceinline__ void async_g2l_b128(const void* gptr, void* lptr) {
    unsigned long long ga = (unsigned long long)gptr;
    unsigned int la = (unsigned int)(unsigned long long)lptr;  // low 32 = LDS byte addr
    asm volatile("global_load_async_to_lds_b128 %0, %1, off"
                 :: "v"(la), "v"(ga) : "memory");
}
__device__ __forceinline__ void wait_async0() {
    asm volatile("s_wait_asynccnt 0x0" ::: "memory");
}

// ============ kernel 1: per-(b,c) sums for x1, x2, x=x1+x2 ============
// sums layout: k*256 + (b*64+c), k = {sum_x1, ssq_x1, sum_x2, ssq_x2, sum_x, ssq_x}
__global__ __launch_bounds__(256) void k_stats(const float* __restrict__ x1,
                                               const float* __restrict__ x2,
                                               float* __restrict__ sums) {
    int bc = blockIdx.x, t = threadIdx.x;
    const float* p1 = x1 + (long)bc * Nn;
    const float* p2 = x2 + (long)bc * Nn;
    float s1 = 0, q1 = 0, s2 = 0, q2 = 0, sx = 0, qx = 0;
    for (int n = t; n < Nn; n += 256) {
        float a = p1[n], b = p2[n], x = a + b;
        s1 += a; q1 += a * a; s2 += b; q2 += b * b; sx += x; qx += x * x;
    }
    __shared__ float red[6][256];
    red[0][t] = s1; red[1][t] = q1; red[2][t] = s2;
    red[3][t] = q2; red[4][t] = sx; red[5][t] = qx;
    __syncthreads();
    for (int s = 128; s > 0; s >>= 1) {
        if (t < s)
            for (int k = 0; k < 6; k++) red[k][t] += red[k][t + s];
        __syncthreads();
    }
    if (t == 0)
        for (int k = 0; k < 6; k++) sums[k * 256 + bc] = red[k][0];
}

// ============ kernel 2: GN stats (3 tensors) + SE gate y ============
// stats layout: [tensor(3)][b*32+g][{mu,rsig}]
__global__ __launch_bounds__(256) void k_prep(const float* __restrict__ sums,
                                              const float* __restrict__ se_w1,
                                              const float* __restrict__ se_w2,
                                              float* __restrict__ stats,
                                              float* __restrict__ yse) {
    int t = threadIdx.x;
    for (int idx = t; idx < 3 * Bsz * Gg; idx += 256) {
        int tz = idx >> 7, bg = idx & 127;
        int b = bg >> 5, g = bg & 31;
        int c0 = b * 64 + g * 2;
        const float* S = sums + (tz * 2 + 0) * 256;
        const float* Q = sums + (tz * 2 + 1) * 256;
        float mean = (S[c0] + S[c0 + 1]) * (1.0f / 8192.0f);
        float ex2  = (Q[c0] + Q[c0 + 1]) * (1.0f / 8192.0f);
        float var = ex2 - mean * mean;
        stats[(tz * 128 + bg) * 2 + 0] = mean;
        stats[(tz * 128 + bg) * 2 + 1] = rsqrtf(var + 1e-6f);
    }
    __shared__ float hid[Bsz * Rr];
    __syncthreads();
    if (t < Bsz * Rr) {
        int b = t >> 2, r = t & 3;
        float acc = 0.f;
        for (int c = 0; c < Cc; c++)
            acc += (sums[4 * 256 + b * 64 + c] * (1.0f / 4096.0f)) * se_w1[r * 64 + c];
        hid[t] = fmaxf(acc, 0.f);
    }
    __syncthreads();
    {
        int b = t >> 6, c = t & 63;
        float acc = 0.f;
        for (int r = 0; r < Rr; r++) acc += hid[b * 4 + r] * se_w2[c * 4 + r];
        yse[t] = sigmoidf_(acc);
    }
}

// ============ kernel 3: GroupNorm + 1x1-conv projections via WMMA ============
// grid (128, B, 3): z=0 x1->q1,k1 ; z=1 x2->q2,k2 ; z=2 (x1+x2)->v
// q/k stored transposed (b,n,c) bf16; v stored natural (b,c,n) bf16
__global__ __launch_bounds__(256) void k_proj(const float* __restrict__ x1,
                                              const float* __restrict__ x2,
                                              const float* __restrict__ gamma,
                                              const float* __restrict__ beta,
                                              const float* __restrict__ wq, const float* __restrict__ bq,
                                              const float* __restrict__ wk, const float* __restrict__ bk,
                                              const float* __restrict__ wv, const float* __restrict__ bv,
                                              const float* __restrict__ stats,
                                              unsigned short* __restrict__ q1o,
                                              unsigned short* __restrict__ k1o,
                                              unsigned short* __restrict__ q2o,
                                              unsigned short* __restrict__ k2o,
                                              unsigned short* __restrict__ vo) {
    const int nbase = blockIdx.x * 32;
    const int b = blockIdx.y;
    const int z = blockIdx.z;
    const int t = threadIdx.x;

    __shared__ __align__(16) unsigned short xin[32][64];     // [n][c] normalized bf16
    __shared__ __align__(16) unsigned short wl[2][64][64];   // bf16 weights
    __shared__ float bl[2][64];

    const int nW = (z == 2) ? 1 : 2;
    const float* Wsrc[2] = { (z == 2) ? wv : wq, wk };
    const float* Bias[2] = { (z == 2) ? bv : bq, bk };
    unsigned short* wf = &wl[0][0][0];
    for (int i = t; i < nW * 4096; i += 256)
        wf[i] = f2bf(Wsrc[i >> 12][i & 4095]);
    for (int i = t; i < nW * 64; i += 256)
        (&bl[0][0])[i] = Bias[i >> 6][i & 63];

    // stage normalized input tile (64c x 32n) transposed into LDS as [n][c]
    for (int i = t; i < 2048; i += 256) {
        int c = i >> 5, n = i & 31;
        long off = ((long)(b * 64 + c)) * Nn + nbase + n;
        float v = (z == 0) ? x1[off] : (z == 1) ? x2[off] : (x1[off] + x2[off]);
        int bg = b * 32 + (c >> 1);
        float mu = stats[(z * 128 + bg) * 2 + 0];
        float rs = stats[(z * 128 + bg) * 2 + 1];
        xin[n][c] = f2bf((v - mu) * rs * gamma[c] + beta[c]);
    }
    __syncthreads();

    const int wave = t >> 5, lane = t & 31, half = lane >> 4, l15 = lane & 15;
    const int mtile = wave & 3, jtile = wave >> 2;   // 4 m-tiles x 2 n-tiles

    // B fragments (xn): K=c = cs*32 + 16*(e>>3)+8*half+(e&7), col n = jtile*16 + l15
    Frag16 Bf[2];
    for (int cs = 0; cs < 2; cs++) {
        const unsigned short* src = &xin[jtile * 16 + l15][cs * 32 + 8 * half];
        Bf[cs].q[0] = *(const uint4v*)(src);
        Bf[cs].q[1] = *(const uint4v*)(src + 16);
    }

    for (int wsel = 0; wsel < nW; wsel++) {
        v8f acc = {0.f, 0.f, 0.f, 0.f, 0.f, 0.f, 0.f, 0.f};
        for (int cs = 0; cs < 2; cs++) {
            // A fragment (weights): row m = mtile*16 + l15, K = 16*(e>>3)+8*half+(e&7)
            Frag16 Af;
            const unsigned short* srcA = &wl[wsel][mtile * 16 + l15][cs * 32 + 8 * half];
            Af.q[0] = *(const uint4v*)(srcA);
            Af.q[1] = *(const uint4v*)(srcA + 16);
            acc = __builtin_amdgcn_wmma_f32_16x16x32_bf16(false, Af.v, false, Bf[cs].v,
                                                          (short)0, acc, false, false);
        }
        for (int r = 0; r < 8; r++)
            acc[r] += bl[wsel][mtile * 16 + 8 * half + r];

        int ng = nbase + jtile * 16 + l15;
        if (z < 2) {
            // transposed store (b,n,c): lane owns col n; rows m contiguous -> 1x b128
            unsigned short* dst = (z == 0) ? (wsel ? k1o : q1o) : (wsel ? k2o : q2o);
            uint4v pk = { f2bf2(acc[0], acc[1]), f2bf2(acc[2], acc[3]),
                          f2bf2(acc[4], acc[5]), f2bf2(acc[6], acc[7]) };
            *(uint4v*)(dst + ((long)(b * Nn) + ng) * 64 + mtile * 16 + 8 * half) = pk;
        } else {
            // natural (b,c,n): 16 lanes of a row write contiguous n
            for (int r = 0; r < 8; r++) {
                int m = mtile * 16 + 8 * half + r;
                vo[((long)(b * 64 + m)) * Nn + ng] = f2bf(acc[r]);
            }
        }
    }
}

// ============ kernel 4: flash attention via WMMA (TJ = 64) ============
// grid (64 i-tiles, B, 2 branches); block = 256 (8 waves)
__global__ __launch_bounds__(256) void k_attn(const unsigned short* __restrict__ qk,
                                              const unsigned short* __restrict__ vbuf,
                                              float* __restrict__ hbuf) {
    const int itb = blockIdx.x, b = blockIdx.y, br = blockIdx.z;
    const unsigned short* Q = qk + (long)br * 2 * SZQ + (long)b * Nn * Cc;
    const unsigned short* K = qk + (long)br * 2 * SZQ + SZQ + (long)b * Nn * Cc;
    const unsigned short* V = vbuf + (long)b * Cc * Nn;
    float* H = hbuf + ((long)br * Bsz + b) * (long)Cc * Nn;

    __shared__ __align__(16) float s_lds[64][64];            // 16 KB
    __shared__ __align__(16) unsigned short p_lds[64][64];   // 8 KB
    __shared__ __align__(16) unsigned short v_lds[64][64];   // 8 KB
    __shared__ float alpha_lds[64];
    __shared__ float l_lds[64];

    const int t = threadIdx.x, wave = t >> 5, lane = t & 31;
    const int half = lane >> 4, l15 = lane & 15;
    const int it = wave & 3;        // i-tile (for S and O)
    const int jp = wave >> 2;       // j-tile pair for S / c-tile pair for O
    const int ibase = itb * 64;
    const int srow = t >> 2, spart = t & 3;   // softmax: 4 threads/row, 16 cols each

    // Q A-fragments (rows i = ibase + it*16 + l15), both c-steps; kept in registers
    Frag16 Qf[2];
    {
        const unsigned short* qrow = Q + ((long)(ibase + it * 16 + l15)) * 64;
        for (int cs = 0; cs < 2; cs++) {
            Qf[cs].q[0] = *(const uint4v*)(qrow + cs * 32 + 8 * half);
            Qf[cs].q[1] = *(const uint4v*)(qrow + cs * 32 + 16 + 8 * half);
        }
    }
    v8f Oacc[2];
    Oacc[0] = (v8f){0.f,0.f,0.f,0.f,0.f,0.f,0.f,0.f};
    Oacc[1] = Oacc[0];
    float m_run = -1e30f, l_run = 0.f;

    for (int jc = 0; jc < 64; jc++) {
        const int jbase = jc * 64;
        // ---- kick off async V chunk (global -> LDS), overlaps with QK^T WMMAs ----
        {
            int c = t >> 2, j16 = (t & 3) * 16;
            async_g2l_b128(V + (long)c * Nn + jbase + j16,     &v_lds[c][j16]);
            async_g2l_b128(V + (long)c * Nn + jbase + j16 + 8, &v_lds[c][j16 + 8]);
        }
        // ---- S tiles: wave computes S[it, jp*2+sub], sub=0,1 ----
        for (int sub = 0; sub < 2; sub++) {
            int jt = jp * 2 + sub;
            v8f s = {0.f,0.f,0.f,0.f,0.f,0.f,0.f,0.f};
            for (int cs = 0; cs < 2; cs++) {
                Frag16 Kf;   // B: col j = jbase + jt*16 + l15 ; K=c contiguous (k is (b,n,c))
                const unsigned short* krow = K + ((long)(jbase + jt * 16 + l15)) * 64
                                               + cs * 32 + 8 * half;
                Kf.q[0] = *(const uint4v*)(krow);
                Kf.q[1] = *(const uint4v*)(krow + 16);
                s = __builtin_amdgcn_wmma_f32_16x16x32_bf16(false, Qf[cs].v, false, Kf.v,
                                                            (short)0, s, false, false);
            }
            for (int r = 0; r < 8; r++)
                s_lds[it * 16 + 8 * half + r][jt * 16 + l15] = s[r] * 0.125f;
        }
        wait_async0();
        __syncthreads();
        // ---- online softmax: 4 threads/row, 16 cols each ----
        {
            f4v sv[4];
            for (int k = 0; k < 4; k++)
                sv[k] = *(const f4v*)(&s_lds[srow][spart * 16 + 4 * k]);
            float mx = -1e30f;
            for (int k = 0; k < 4; k++)
                mx = fmaxf(mx, fmaxf(fmaxf(sv[k].x, sv[k].y), fmaxf(sv[k].z, sv[k].w)));
            mx = fmaxf(mx, __shfl_xor(mx, 1, 32));
            mx = fmaxf(mx, __shfl_xor(mx, 2, 32));
            float m_new = fmaxf(m_run, mx);
            float al = __expf(m_run - m_new);
            float p[16];
            for (int k = 0; k < 4; k++) {
                p[4*k+0] = __expf(sv[k].x - m_new);
                p[4*k+1] = __expf(sv[k].y - m_new);
                p[4*k+2] = __expf(sv[k].z - m_new);
                p[4*k+3] = __expf(sv[k].w - m_new);
            }
            float ps = 0.f;
            for (int k = 0; k < 16; k++) ps += p[k];
            uint4v pk0 = { f2bf2(p[0],p[1]),  f2bf2(p[2],p[3]),
                           f2bf2(p[4],p[5]),  f2bf2(p[6],p[7]) };
            uint4v pk1 = { f2bf2(p[8],p[9]),  f2bf2(p[10],p[11]),
                           f2bf2(p[12],p[13]),f2bf2(p[14],p[15]) };
            *(uint4v*)(&p_lds[srow][spart * 16])     = pk0;
            *(uint4v*)(&p_lds[srow][spart * 16 + 8]) = pk1;
            ps += __shfl_xor(ps, 1, 32);
            ps += __shfl_xor(ps, 2, 32);
            l_run = l_run * al + ps;
            m_run = m_new;
            if (spart == 0) alpha_lds[srow] = al;
        }
        __syncthreads();
        // ---- rescale O, then O += P x V^T over 2 k-steps ----
        {
            float av[8];
            for (int r = 0; r < 8; r++) av[r] = alpha_lds[it * 16 + 8 * half + r];
            for (int r = 0; r < 8; r++) { Oacc[0][r] *= av[r]; Oacc[1][r] *= av[r]; }
            for (int ks = 0; ks < 2; ks++) {
                Frag16 Pf;   // A: row i local, K = j = ks*32 + (16*(e>>3)+8*half+(e&7))
                const unsigned short* prow = &p_lds[it * 16 + l15][ks * 32];
                Pf.q[0] = *(const uint4v*)(prow + 8 * half);
                Pf.q[1] = *(const uint4v*)(prow + 16 + 8 * half);
                for (int cc = 0; cc < 2; cc++) {
                    int ct = jp * 2 + cc;
                    Frag16 Vf;   // B: col c = ct*16+l15, K = j = ks*32 + (16*(e>>3)+8*half+(e&7))
                    const unsigned short* vrow = &v_lds[ct * 16 + l15][ks * 32 + 8 * half];
                    Vf.q[0] = *(const uint4v*)(vrow);
                    Vf.q[1] = *(const uint4v*)(vrow + 16);
                    Oacc[cc] = __builtin_amdgcn_wmma_f32_16x16x32_bf16(false, Pf.v, false, Vf.v,
                                                                       (short)0, Oacc[cc], false, false);
                }
            }
        }
        __syncthreads();
    }
    // ---- finalize: divide by l, store H (b,c,n) fp32 ----
    if (spart == 0) l_lds[srow] = l_run;
    __syncthreads();
    {
        float linv[8];
        for (int r = 0; r < 8; r++) linv[r] = 1.0f / l_lds[it * 16 + 8 * half + r];
        for (int cc = 0; cc < 2; cc++) {
            int c = (jp * 2 + cc) * 16 + l15;
            float* dst = H + (long)c * Nn + ibase + it * 16 + 8 * half;
            f4v o0 = { Oacc[cc][0] * linv[0], Oacc[cc][1] * linv[1],
                       Oacc[cc][2] * linv[2], Oacc[cc][3] * linv[3] };
            f4v o1 = { Oacc[cc][4] * linv[4], Oacc[cc][5] * linv[5],
                       Oacc[cc][6] * linv[6], Oacc[cc][7] * linv[7] };
            *(f4v*)(dst)     = o0;
            *(f4v*)(dst + 4) = o1;
        }
    }
}

// ============ kernel 5: combine 2*h1*sig(x*y) + 2*h2*(1-sig(x*y)) ============
__global__ __launch_bounds__(256) void k_combine(const float* __restrict__ x1,
                                                 const float* __restrict__ x2,
                                                 const float* __restrict__ yse,
                                                 const float* __restrict__ hbuf,
                                                 float* __restrict__ out) {
    long i4 = (long)blockIdx.x * 256 + threadIdx.x;
    long base = i4 * 4;
    int bc = (int)(base >> 12);
    float y = yse[bc];
    f4v a  = *(const f4v*)(x1 + base);
    f4v bb = *(const f4v*)(x2 + base);
    f4v h1 = *(const f4v*)(hbuf + base);
    f4v h2 = *(const f4v*)(hbuf + (long)Bsz * Cc * Nn + base);
    f4v o;
    {
        float cw;
        cw = sigmoidf_((a.x + bb.x) * y); o.x = 2.f * (h1.x * cw + h2.x * (1.f - cw));
        cw = sigmoidf_((a.y + bb.y) * y); o.y = 2.f * (h1.y * cw + h2.y * (1.f - cw));
        cw = sigmoidf_((a.z + bb.z) * y); o.z = 2.f * (h1.z * cw + h2.z * (1.f - cw));
        cw = sigmoidf_((a.w + bb.w) * y); o.w = 2.f * (h1.w * cw + h2.w * (1.f - cw));
    }
    *(f4v*)(out + base) = o;
}

// ============ launch ============
extern "C" void kernel_launch(void* const* d_in, const int* in_sizes, int n_in,
                              void* d_out, int out_size, void* d_ws, size_t ws_size,
                              hipStream_t stream) {
    const float* x1    = (const float*)d_in[0];
    const float* x2    = (const float*)d_in[1];
    const float* se_w1 = (const float*)d_in[2];
    const float* se_w2 = (const float*)d_in[3];
    const float* gamma = (const float*)d_in[4];
    const float* beta  = (const float*)d_in[5];
    const float* wq    = (const float*)d_in[6];
    const float* bq    = (const float*)d_in[7];
    const float* wk    = (const float*)d_in[8];
    const float* bk    = (const float*)d_in[9];
    const float* wv    = (const float*)d_in[10];
    const float* bv    = (const float*)d_in[11];

    char* ws = (char*)d_ws;
    float* sums  = (float*)(ws + 0);                 // 6*256 f32
    float* stats = (float*)(ws + 8192);              // 3*128*2 f32
    float* yse   = (float*)(ws + 12288);             // 256 f32
    unsigned short* qk = (unsigned short*)(ws + 16384);          // q1,k1,q2,k2 (4*2MB)
    unsigned short* vb = qk + 4ll * SZQ;                         // v (2MB)
    float* hbuf = (float*)(ws + 16384 + 10ll * SZQ * 2);         // h1,h2 (8MB f32)

    k_stats<<<256, 256, 0, stream>>>(x1, x2, sums);
    k_prep<<<1, 256, 0, stream>>>(sums, se_w1, se_w2, stats, yse);
    k_proj<<<dim3(128, Bsz, 3), 256, 0, stream>>>(x1, x2, gamma, beta,
                                                  wq, bq, wk, bk, wv, bv, stats,
                                                  qk, qk + SZQ, qk + 2ll * SZQ, qk + 3ll * SZQ, vb);
    k_attn<<<dim3(64, Bsz, 2), 256, 0, stream>>>(qk, vb, hbuf);
    k_combine<<<1024, 256, 0, stream>>>(x1, x2, yse, hbuf, (float*)d_out);
}
